// Joiner_65859028517441
// MI455X (gfx1250) — matrix operand order repaired and checked by reading, not compiled, not run on hardware
//
#include <hip/hip_runtime.h>

// ---------------------------------------------------------------------------
// RNN-T Joiner for MI455X (gfx1250, wave32, WMMA + Tensor Data Mover).
// logits = tanh(hf[b,t,:]+hg[b,u,:]) @ w2^T
//   M = B*T*U = 131072, N = 500 (pad 512), K = 256  -> 33.6 GFLOP (f16 WMMA)
// Output = 262 MB -> ~11 us at 23.3 TB/s HBM: store-bound target.
// TDM (tensor_load_to_lds) stages each 8 KB w2 N-panel in an LDS ring shared
// by 8 waves; each wave owns TWO M-tiles (afrag[2][8], 128 VGPRs) so every
// ds_load_b128 B-fragment feeds two independent WMMA chains -> LDS traffic
// per FLOP halved and ds->wmma latency hidden by dual accumulators + one-
// fragment register lookahead.
// ---------------------------------------------------------------------------

typedef __attribute__((ext_vector_type(16))) _Float16 v16h;
typedef __attribute__((ext_vector_type(8)))  float    v8f;
typedef __attribute__((ext_vector_type(4)))  float    v4f;
typedef __attribute__((ext_vector_type(4)))  unsigned int u32x4;
typedef __attribute__((ext_vector_type(8)))  int      i32x8;
typedef __attribute__((ext_vector_type(4)))  int      i32x4;

#define BN  8
#define TN  256
#define UN  64
#define ENCN 128
#define PREDN 256
#define JD  256      // joiner_dim (K of big GEMM)
#define VOC 500
#define VOCP 512     // padded vocab
#define W1LD 384     // ENC+PRED
#define NT  (VOCP / 16)      // 32 N-tiles
#define PANEL_H 4096         // halves per N-panel (8 K-steps * 512)
#define PANEL_B 8192         // bytes per N-panel

__device__ __forceinline__ float fast_tanh(float x) {
    x = fminf(fmaxf(x, -10.f), 10.f);
    float e = __expf(2.f * x);                 // v_exp_f32 (TRANS, co-executes with WMMA)
    float inv = __builtin_amdgcn_rcpf(e + 1.f);
    return (e - 1.f) * inv;
}

// TDM: copy one contiguous 8 KB panel (2048 dwords) from global to LDS.
// D# per CDNA5 ISA sec 8.3/8.4: group0 = {count=1 | lds_addr | global_addr |
// type=2}; group1 = {data_size=4B, tensor_dim0=64Ki dwords, tensor_dim1=1,
// tile_dim0=2048, tile_dim1=1, tensor_dim0_stride=64Ki}. Remaining groups
// zero (tile <=2-D). Tracked by TENSORcnt.
__device__ __forceinline__ void tdm_load_panel(const _Float16* gsrc, unsigned int lds_off) {
    unsigned long long ga = (unsigned long long)(uintptr_t)gsrc;
    u32x4 g0;
    g0[0] = 1u;                                   // count=1 (valid user D#)
    g0[1] = lds_off;                              // lds_addr [63:32]
    g0[2] = (unsigned int)ga;                     // global_addr[31:0]
    g0[3] = (unsigned int)(ga >> 32) | (2u << 30);// global_addr[56:32] | type=2
    i32x8 g1;
    g1[0] = (int)(2u << 16);        // workgroup_mask=0, data_size=2 (4 B)
    g1[1] = 0;                      // atomic_barrier_addr=0, tensor_dim0[15:0]
    g1[2] = 0x00010001;             // tensor_dim0[31:16]=1 (65536), tensor_dim1[15:0]=1
    g1[3] = (int)(2048u << 16);     // tensor_dim1[31:16]=0, tile_dim0=2048
    g1[4] = 1;                      // tile_dim1=1, tile_dim2=0
    g1[5] = 0x00010000;             // tensor_dim0_stride[31:0] = 65536
    g1[6] = 0;
    g1[7] = 0;
    i32x4 gz4 = {0, 0, 0, 0};
    i32x8 gz8 = {0, 0, 0, 0, 0, 0, 0, 0};
    __builtin_amdgcn_tensor_load_to_lds(g0, g1, gz4, gz4, gz8, 0);
}

// ---------------------------------------------------------------------------
// Prep: hf[b*T+t][j] = dot(ft[b,t,:128], w1[j,0:128])
//       hg[b*U+u][j] = dot(gu[b,u,:256], w1[j,128:384])
// ---------------------------------------------------------------------------
#define RPB 8
__global__ void __launch_bounds__(256) joiner_prep(
    const float* __restrict__ ft, const float* __restrict__ gu,
    const float* __restrict__ w1, float* __restrict__ hf, float* __restrict__ hg)
{
    __shared__ float xs[RPB * PREDN];          // 8 KB max
    const int j = threadIdx.x;                 // output channel 0..255
    const int nfBlocks = (BN * TN) / RPB;      // 256
    int K; const float* src; const float* wb; float* dst;
    if ((int)blockIdx.x < nfBlocks) {
        int row0 = blockIdx.x * RPB;
        K = ENCN;  src = ft + row0 * ENCN;  wb = w1;         dst = hf + row0 * JD;
    } else {
        int row0 = (blockIdx.x - nfBlocks) * RPB;
        K = PREDN; src = gu + row0 * PREDN; wb = w1 + ENCN;  dst = hg + row0 * JD;
    }
    for (int i = threadIdx.x; i < RPB * K; i += 256) xs[i] = src[i];
    __syncthreads();

    float acc[RPB];
#pragma unroll
    for (int r = 0; r < RPB; ++r) acc[r] = 0.f;
    for (int e = 0; e < K; ++e) {
        float wv = wb[j * W1LD + e];
#pragma unroll
        for (int r = 0; r < RPB; ++r) acc[r] = fmaf(xs[r * K + e], wv, acc[r]);
    }
#pragma unroll
    for (int r = 0; r < RPB; ++r) dst[r * JD + j] = acc[r];
}

// ---------------------------------------------------------------------------
// Pack w2 (500x256 f32) -> f16 in exact WMMA B-fragment order:
//   flat = ((n*8+s)*32 + lane)*16 + jj
//   lane 0-15 : N = n*16+lane,    K = 32s + (jj<8 ? jj : 8+jj)
//   lane16-31 : N = n*16+lane-16, K = 32s + 8 + (jj<8 ? jj : 8+jj)
// ---------------------------------------------------------------------------
__global__ void __launch_bounds__(256) joiner_pack_w2(
    const float* __restrict__ w2, _Float16* __restrict__ w2p)
{
    int idx  = blockIdx.x * 256 + threadIdx.x;     // 0 .. 512*256-1
    int jj   = idx & 15;
    int lane = (idx >> 4) & 31;
    int sn   = idx >> 9;                           // n*8 + s
    int s    = sn & 7, n = sn >> 3;
    int v    = n * 16 + (lane & 15);
    int k    = 32 * s + ((lane >> 4) << 3) + jj + ((jj >> 3) << 3);
    float val = (v < VOC) ? w2[v * JD + k] : 0.f;
    w2p[idx] = (_Float16)val;
}

// ---------------------------------------------------------------------------
// Main: one wave per PAIR of M-tiles = (b, t, u0..u0+31); 8 waves/block.
// A tiles (2 x 16x256 f16) resident in afrag[2][8] (128 VGPRs), ISA 16-bit
// A layout. B panels TDM-staged in a 3-buffer LDS ring shared by all waves;
// each B fragment read feeds both accumulator chains.
// ---------------------------------------------------------------------------
__global__ void __launch_bounds__(256) joiner_main(
    const float* __restrict__ hf, const float* __restrict__ hg,
    const _Float16* __restrict__ w2p, float* __restrict__ out)
{
    __shared__ _Float16 bpanel[3][PANEL_H];        // 3 x 8 KB ring

    const int wave = threadIdx.x >> 5;
    const int lane = threadIdx.x & 31;
    const int wt   = blockIdx.x * 8 + wave;        // 0 .. 4095
    const int utp  = wt & 1;                       // u-pair: u0 = 32*utp
    const int bt   = wt >> 1;                      // b*T + t
    const int b    = bt >> 8;
    const int u0   = utp << 5;
    const int m    = lane & 15;
    const int half = lane >> 4;

    const unsigned int lds_base = (unsigned int)(uintptr_t)(void*)&bpanel[0][0];

    // Prefetch panels 0 and 1 via TDM before the (long) A build.
    if (wave == 0) {
        tdm_load_panel(w2p,            lds_base);
        tdm_load_panel(w2p + PANEL_H,  lds_base + PANEL_B);
    }

    const float* hfp  = hf + bt * JD;
    const float* hgp0 = hg + (b * UN + u0 + m) * JD;        // rows u0+m
    const float* hgp1 = hgp0 + 16 * JD;                     // rows u0+16+m

    // Build both 16x256 A operands in registers; hf row loads shared.
    v16h afrag0[8], afrag1[8];
#pragma unroll
    for (int s = 0; s < 8; ++s) {
        const int kb = 32 * s + half * 8;
        v4f f0 = *(const v4f*)(hfp + kb);
        v4f f1 = *(const v4f*)(hfp + kb + 4);
        v4f f2 = *(const v4f*)(hfp + kb + 16);
        v4f f3 = *(const v4f*)(hfp + kb + 20);
        v4f g0 = *(const v4f*)(hgp0 + kb);
        v4f g1 = *(const v4f*)(hgp0 + kb + 4);
        v4f g2 = *(const v4f*)(hgp0 + kb + 16);
        v4f g3 = *(const v4f*)(hgp0 + kb + 20);
        v4f h0 = *(const v4f*)(hgp1 + kb);
        v4f h1 = *(const v4f*)(hgp1 + kb + 4);
        v4f h2 = *(const v4f*)(hgp1 + kb + 16);
        v4f h3 = *(const v4f*)(hgp1 + kb + 20);
#pragma unroll
        for (int j = 0; j < 4; ++j) {
            afrag0[s][j]      = (_Float16)fast_tanh(f0[j] + g0[j]);
            afrag0[s][4 + j]  = (_Float16)fast_tanh(f1[j] + g1[j]);
            afrag0[s][8 + j]  = (_Float16)fast_tanh(f2[j] + g2[j]);
            afrag0[s][12 + j] = (_Float16)fast_tanh(f3[j] + g3[j]);
            afrag1[s][j]      = (_Float16)fast_tanh(f0[j] + h0[j]);
            afrag1[s][4 + j]  = (_Float16)fast_tanh(f1[j] + h1[j]);
            afrag1[s][8 + j]  = (_Float16)fast_tanh(f2[j] + h2[j]);
            afrag1[s][12 + j] = (_Float16)fast_tanh(f3[j] + h3[j]);
        }
    }

    float* obase = out + ((size_t)(bt * UN + u0)) * VOC;

    for (int n = 0; n < NT; ++n) {
        // Panel n must have landed; panel n+1 may still be in flight.
        if (wave == 0) {
            if (n < NT - 1) __builtin_amdgcn_s_wait_tensorcnt(1);
            else            __builtin_amdgcn_s_wait_tensorcnt(0);
        }
        // Publishes panel n to all waves AND proves every wave finished
        // reading buffer (n+2)%3 (last read in iteration n-1).
        __syncthreads();
        if (wave == 0 && n + 2 < NT) {
            int nb = (n + 2) % 3;
            tdm_load_panel(w2p + (size_t)(n + 2) * PANEL_H,
                           lds_base + (unsigned int)nb * PANEL_B);
        }

        const _Float16* pb = &bpanel[n % 3][lane * 16];
        v8f acc0 = {}, acc1 = {};
        v16h bcur = *(const v16h*)(pb);              // 2x ds_load_b128
#pragma unroll
        for (int s = 0; s < 8; ++s) {
            v16h bnxt = bcur;
            if (s < 7) bnxt = *(const v16h*)(pb + (s + 1) * 512);  // lookahead
            acc0 = __builtin_amdgcn_wmma_f32_16x16x32_f16(
                false, afrag0[s], false, bcur, (short)0, acc0, false, false);
            acc1 = __builtin_amdgcn_wmma_f32_16x16x32_f16(
                false, afrag1[s], false, bcur, (short)0, acc1, false, false);
            bcur = bnxt;
        }
        // C/D layout: VGPR r -> row M = r + 8*half, col N = m (within tile).
        const int v = n * 16 + m;
        if (v < VOC) {
            float* orow0 = obase + (half * 8) * VOC + v;
            float* orow1 = orow0 + 16 * VOC;
#pragma unroll
            for (int r = 0; r < 8; ++r) orow0[r * VOC] = acc0[r];
#pragma unroll
            for (int r = 0; r < 8; ++r) orow1[r * VOC] = acc1[r];
        }
    }
}

extern "C" void kernel_launch(void* const* d_in, const int* in_sizes, int n_in,
                              void* d_out, int out_size, void* d_ws, size_t ws_size,
                              hipStream_t stream) {
    const float* ft = (const float*)d_in[0];   // (8,256,128)
    const float* gu = (const float*)d_in[1];   // (8,64,256)
    const float* w1 = (const float*)d_in[2];   // (256,384)
    const float* w2 = (const float*)d_in[3];   // (500,256)
    float* out = (float*)d_out;                // (8,256,64,500)

    // Workspace: [w2p f16 512*256 = 256KB][hf f32 2048*256 = 2MB][hg f32 512*256 = 512KB]
    _Float16* w2p = (_Float16*)d_ws;
    float* hf = (float*)((char*)d_ws + (size_t)VOCP * JD * sizeof(_Float16));
    float* hg = hf + (size_t)BN * TN * JD;

    joiner_prep<<<(BN * TN + BN * UN) / RPB, 256, 0, stream>>>(ft, gu, w1, hf, hg);
    joiner_pack_w2<<<(VOCP * JD) / 256, 256, 0, stream>>>(w2, w2p);
    joiner_main<<<(BN * TN * (UN / 32)) / 8, 256, 0, stream>>>(hf, hg, w2p, out);
}
